// MultiHeadAttention_70033736729149
// MI455X (gfx1250) — matrix operand order repaired and checked by reading, not compiled, not run on hardware
//
#include <hip/hip_runtime.h>
#include <math.h>

typedef _Float16 f16;
typedef _Float16 v16h __attribute__((ext_vector_type(16)));
typedef _Float16 v8h  __attribute__((ext_vector_type(8)));
typedef _Float16 v4h  __attribute__((ext_vector_type(4)));
typedef float    v8f  __attribute__((ext_vector_type(8)));

#define DEVI __device__ __forceinline__

constexpr int Bc = 2, Tc = 2048, Dc = 2048, Hc = 16, KVc = 4, HDc = 128;

union Frag16 { v16h v; v8h h[2]; };

DEVI v8f wmma_f16(const v16h& a, const v16h& b, const v8f& c) {
  // D = A(16x32 f16) * B(32x16 f16) + C(16x16 f32)
  return __builtin_amdgcn_wmma_f32_16x16x32_f16(false, a, false, b, (short)0, c, false, false);
}

// Async global -> LDS copy of 16 bytes per lane (ASYNCcnt-tracked, CDNA5).
DEVI void async_copy_b128(unsigned lds_off, const f16* gptr) {
  asm volatile("global_load_async_to_lds_b128 %0, %1, off"
               :: "v"(lds_off), "v"((unsigned long long)(size_t)gptr)
               : "memory");
}
DEVI void wait_async0() { asm volatile("s_wait_asynccnt 0" ::: "memory"); }

// ---------------------------------------------------------------------------
// fp32 -> f16 conversion, 4 elements / thread
// ---------------------------------------------------------------------------
__global__ void cvt_f32_f16(const float4* __restrict__ in, v4h* __restrict__ out, int n4) {
  int i = blockIdx.x * blockDim.x + threadIdx.x;
  if (i < n4) {
    float4 v = in[i];
    v4h o; o[0] = (f16)v.x; o[1] = (f16)v.y; o[2] = (f16)v.z; o[3] = (f16)v.w;
    out[i] = o;
  }
}

// ---------------------------------------------------------------------------
// WMMA GEMM: C[M,N] = A[M,K](f16 row-major) * Bw[K,N](f16 row-major)
// 256 threads = 8 waves (4x2); block tile 128x128; wave tile 32x64; K-step 32.
// A tile staged with global_load_async_to_lds_b128; B tile staged transposed.
// MODE 0: out f16 at ((b*NH+h)*T + t)*HD + d        (Q / K head-major)
// MODE 1: out f16 at ((b*NH+h)*HD + d)*T + t        (V transposed)
// MODE 2: out f32 row-major                          (final y)
// ---------------------------------------------------------------------------
template <int MODE>
__global__ __launch_bounds__(256) void gemm_wmma(
    const f16* __restrict__ A, const f16* __restrict__ Bw, void* __restrict__ outp,
    int M, int N, int K, int NH) {
  __shared__ f16 As[128][40];   // 32 used + 8 pad; row stride 80B (16B aligned)
  __shared__ f16 Bs[128][40];   // B^T tile: Bs[n][k]

  const int tid  = threadIdx.x;
  const int lane = tid & 31;
  const int wid  = tid >> 5;
  const int wm = wid & 3;       // wave row   (4 x 32 rows)
  const int wn = wid >> 2;      // wave col   (2 x 64 cols)
  const int hl = lane >> 4;     // half of wave (0/1)
  const int ln = lane & 15;
  const int m0 = blockIdx.y * 128;
  const int n0 = blockIdx.x * 128;

  v8f acc[2][4] = {};

  // staging coordinates: A, 16 halves per thread (two b128 async copies)
  const int am = tid >> 1;
  const int ak = (tid & 1) * 16;
  const unsigned lds_a0 = (unsigned)(size_t)&As[am][ak];
  const unsigned lds_a1 = (unsigned)(size_t)&As[am][ak + 8];
  // B: 32 k-rows, 8 contiguous n each, scattered transposed
  const int bk = tid >> 3;
  const int bn = (tid & 7) * 8;

  for (int k0 = 0; k0 < K; k0 += 32) {
    // A tile: global -> LDS via async DMA path (no VGPR round trip)
    {
      const f16* ga = &A[(size_t)(m0 + am) * K + (k0 + ak)];
      async_copy_b128(lds_a0, ga);
      async_copy_b128(lds_a1, ga + 8);
    }
    // B tile transposed: read 8 contiguous n, scatter into Bs[n][k]
#pragma unroll
    for (int it = 0; it < 2; ++it) {
      int n = it * 64 + bn;
      v8h w = *(const v8h*)&Bw[(size_t)(k0 + bk) * N + (n0 + n)];
#pragma unroll
      for (int j = 0; j < 8; ++j) Bs[n + j][bk] = w[j];
    }
    wait_async0();
    __syncthreads();

    Frag16 a[2], b[4];
#pragma unroll
    for (int mt = 0; mt < 2; ++mt) {
      int m = wm * 32 + mt * 16 + ln;
      a[mt].h[0] = *(const v8h*)&As[m][hl * 8];
      a[mt].h[1] = *(const v8h*)&As[m][16 + hl * 8];
    }
#pragma unroll
    for (int nt = 0; nt < 4; ++nt) {
      int n = wn * 64 + nt * 16 + ln;
      b[nt].h[0] = *(const v8h*)&Bs[n][hl * 8];
      b[nt].h[1] = *(const v8h*)&Bs[n][16 + hl * 8];
    }
#pragma unroll
    for (int mt = 0; mt < 2; ++mt)
#pragma unroll
      for (int nt = 0; nt < 4; ++nt)
        acc[mt][nt] = wmma_f16(a[mt].v, b[nt].v, acc[mt][nt]);
    __syncthreads();
  }

  // epilogue: C layout lane holds col n=l&15, VGPR r holds row r + 8*(l>>4)
#pragma unroll
  for (int mt = 0; mt < 2; ++mt)
#pragma unroll
    for (int nt = 0; nt < 4; ++nt)
#pragma unroll
      for (int r = 0; r < 8; ++r) {
        int row = m0 + wm * 32 + mt * 16 + r + hl * 8;
        int col = n0 + wn * 64 + nt * 16 + ln;
        float val = acc[mt][nt][r];
        if (MODE == 2) {
          ((float*)outp)[(size_t)row * N + col] = val;
        } else {
          int bb = row / Tc, t = row % Tc;
          int h  = col / HDc, d = col % HDc;
          f16* o = (f16*)outp;
          if (MODE == 0)
            o[(((size_t)(bb * NH + h)) * Tc + t) * HDc + d] = (f16)val;
          else
            o[(((size_t)(bb * NH + h)) * HDc + d) * Tc + t] = (f16)val;
        }
      }
}

// ---------------------------------------------------------------------------
// RoPE on (B*NH, T, HD) f16, interleaved even/odd pairs
// ---------------------------------------------------------------------------
__global__ void rope_kernel(f16* __restrict__ qk, int total_pairs) {
  int idx = blockIdx.x * blockDim.x + threadIdx.x;
  if (idx >= total_pairs) return;
  int j  = idx % (HDc / 2);
  int t  = (idx / (HDc / 2)) % Tc;
  int bh = idx / ((HDc / 2) * Tc);
  const float LOGB_OVER_HD = 9.2103403719761836f / (float)HDc;  // ln(10000)/HD
  float ang = (float)t * __expf(-(float)(2 * j) * LOGB_OVER_HD);
  float s, c;
  __sincosf(ang, &s, &c);
  size_t base = (((size_t)bh * Tc) + t) * HDc + 2 * j;
  float x1 = (float)qk[base], x2 = (float)qk[base + 1];
  qk[base]     = (f16)(x1 * c - x2 * s);
  qk[base + 1] = (f16)(x1 * s + x2 * c);
}

// ---------------------------------------------------------------------------
// Causal GQA flash attention.
// One wave per 16-row q tile. 32-key chunks: Q.K^T (8 wmma) + P.V (8 wmma).
// Q(B,H,T,HD), K(B,KV,T,HD) [rows feed B-frag of QK^T directly],
// Vt(B,KV,HD,T) [rows feed B-frag of P.V directly], O(B,T,H*HD) f16.
// ---------------------------------------------------------------------------
__global__ __launch_bounds__(128) void attn_kernel(
    const f16* __restrict__ Q, const f16* __restrict__ Kh,
    const f16* __restrict__ Vt, f16* __restrict__ O) {
  __shared__ f16 Ps[4][16][40];  // per-wave P bounce (C-layout -> A-layout)

  const int wid  = threadIdx.x >> 5;
  const int lane = threadIdx.x & 31;
  const int hl = lane >> 4, ln = lane & 15;
  const int h = blockIdx.y, b = blockIdx.z;
  const int q0 = (blockIdx.x * 4 + wid) * 16;
  const int kv = h / (Hc / KVc);

  const f16* qbase = Q  + ((size_t)(b * Hc  + h))  * Tc * HDc;
  const f16* kbase = Kh + ((size_t)(b * KVc + kv)) * Tc * HDc;
  const f16* vbase = Vt + ((size_t)(b * KVc + kv)) * HDc * Tc;

  // Q fragments for 4 K-steps of 32 along HD
  Frag16 qf[4];
  {
    const f16* qrow = qbase + (size_t)(q0 + ln) * HDc;
#pragma unroll
    for (int kd = 0; kd < 4; ++kd) {
      qf[kd].h[0] = *(const v8h*)&qrow[kd * 32 + hl * 8];
      qf[kd].h[1] = *(const v8h*)&qrow[kd * 32 + 16 + hl * 8];
    }
  }

  v8f o[8] = {};
  float run_m[8], run_s[8];
#pragma unroll
  for (int r = 0; r < 8; ++r) { run_m[r] = -3.0e38f; run_s[r] = 0.0f; }
  const float scale = 0.08838834764831845f;  // 1/sqrt(128)

  for (int c0 = 0; c0 <= q0 + 15; c0 += 32) {
    // S = Q K^T for two 16-key tiles
    v8f s[2] = {};
#pragma unroll
    for (int kt = 0; kt < 2; ++kt) {
      const f16* krow = kbase + (size_t)(c0 + kt * 16 + ln) * HDc;
#pragma unroll
      for (int kd = 0; kd < 4; ++kd) {
        Frag16 kf;
        kf.h[0] = *(const v8h*)&krow[kd * 32 + hl * 8];
        kf.h[1] = *(const v8h*)&krow[kd * 32 + 16 + hl * 8];
        s[kt] = wmma_f16(qf[kd].v, kf.v, s[kt]);
      }
    }
    // scale + causal mask (VALU select, no divergence)
    const bool need_mask = (c0 + 31 > q0);
#pragma unroll
    for (int kt = 0; kt < 2; ++kt)
#pragma unroll
      for (int r = 0; r < 8; ++r) {
        float v = s[kt][r] * scale;
        if (need_mask) {
          int key = c0 + kt * 16 + ln;
          int qq  = q0 + r + hl * 8;
          if (key > qq) v = -3.0e38f;
        }
        s[kt][r] = v;
      }
    // online softmax: rows live across 16 lanes of each wave half
    float alpha[8];
#pragma unroll
    for (int r = 0; r < 8; ++r) {
      float m = fmaxf(s[0][r], s[1][r]);
#pragma unroll
      for (int off = 1; off <= 8; off <<= 1) m = fmaxf(m, __shfl_xor(m, off, 32));
      float nm = fmaxf(run_m[r], m);
      alpha[r] = __expf(run_m[r] - nm);
      run_m[r] = nm;
      float p0 = __expf(s[0][r] - nm);
      float p1 = __expf(s[1][r] - nm);
      s[0][r] = p0; s[1][r] = p1;
      float ps = p0 + p1;
#pragma unroll
      for (int off = 1; off <= 8; off <<= 1) ps += __shfl_xor(ps, off, 32);
      run_s[r] = run_s[r] * alpha[r] + ps;
    }
#pragma unroll
    for (int nt = 0; nt < 8; ++nt)
#pragma unroll
      for (int r = 0; r < 8; ++r) o[nt][r] *= alpha[r];

    // P: C-layout -> LDS -> A-layout (DS in-order per wave; fence load/store)
#pragma unroll
    for (int kt = 0; kt < 2; ++kt)
#pragma unroll
      for (int r = 0; r < 8; ++r)
        Ps[wid][r + hl * 8][kt * 16 + ln] = (f16)s[kt][r];
    asm volatile("s_wait_dscnt 0" ::: "memory");
    Frag16 pf;
    pf.h[0] = *(const v8h*)&Ps[wid][ln][hl * 8];
    pf.h[1] = *(const v8h*)&Ps[wid][ln][16 + hl * 8];

    // O += P V  (V^T rows feed B-frags directly)
#pragma unroll
    for (int nt = 0; nt < 8; ++nt) {
      const f16* vrow = vbase + (size_t)(nt * 16 + ln) * Tc + c0;
      Frag16 vf;
      vf.h[0] = *(const v8h*)&vrow[hl * 8];
      vf.h[1] = *(const v8h*)&vrow[16 + hl * 8];
      o[nt] = wmma_f16(pf.v, vf.v, o[nt]);
    }
  }

  // finalize rows, store O as (b, t, h*HD + d) so it is GEMM-A-ready
#pragma unroll
  for (int r = 0; r < 8; ++r) {
    float inv = 1.0f / run_s[r];
    int t = q0 + r + hl * 8;
    f16* orow = O + (((size_t)b * Tc) + t) * (Hc * HDc) + h * HDc;
#pragma unroll
    for (int nt = 0; nt < 8; ++nt)
      orow[nt * 16 + ln] = (f16)(o[nt][r] * inv);
  }
}

// ---------------------------------------------------------------------------
extern "C" void kernel_launch(void* const* d_in, const int* in_sizes, int n_in,
                              void* d_out, int out_size, void* d_ws, size_t ws_size,
                              hipStream_t stream) {
  (void)in_sizes; (void)n_in; (void)out_size; (void)ws_size;
  const float* x  = (const float*)d_in[0];
  const float* Wq = (const float*)d_in[1];
  const float* Wk = (const float*)d_in[2];
  const float* Wv = (const float*)d_in[3];
  const float* Wo = (const float*)d_in[4];
  float* y = (float*)d_out;

  char* ws = (char*)d_ws;
  size_t off = 0;
  auto alloc = [&](size_t nhalf) {
    f16* p = (f16*)(ws + off);
    off = (off + nhalf * sizeof(f16) + 255) & ~(size_t)255;
    return p;
  };
  f16* xh  = alloc((size_t)Bc * Tc * Dc);
  f16* Wqh = alloc((size_t)Dc * Hc * HDc);
  f16* Wkh = alloc((size_t)Dc * KVc * HDc);
  f16* Wvh = alloc((size_t)Dc * KVc * HDc);
  f16* Woh = alloc((size_t)Hc * HDc * Dc);
  f16* Qh  = alloc((size_t)Bc * Hc * Tc * HDc);
  f16* Kh  = alloc((size_t)Bc * KVc * Tc * HDc);
  f16* Vt  = alloc((size_t)Bc * KVc * HDc * Tc);
  f16* Oh  = alloc((size_t)Bc * Tc * Hc * HDc);

  auto cvt = [&](const float* in, f16* out, int n) {
    int n4 = n / 4;
    cvt_f32_f16<<<(n4 + 255) / 256, 256, 0, stream>>>((const float4*)in, (v4h*)out, n4);
  };
  cvt(x,  xh,  Bc * Tc * Dc);
  cvt(Wq, Wqh, Dc * Hc * HDc);
  cvt(Wk, Wkh, Dc * KVc * HDc);
  cvt(Wv, Wvh, Dc * KVc * HDc);
  cvt(Wo, Woh, Hc * HDc * Dc);

  const int M = Bc * Tc;  // 4096
  dim3 blk(256);
  // Q = x@Wq -> (B,H,T,HD); K = x@Wk -> (B,KV,T,HD); V^T = (x@Wv)^T -> (B,KV,HD,T)
  gemm_wmma<0><<<dim3((Hc * HDc) / 128, M / 128), blk, 0, stream>>>(xh, Wqh, Qh, M, Hc * HDc, Dc, Hc);
  gemm_wmma<0><<<dim3((KVc * HDc) / 128, M / 128), blk, 0, stream>>>(xh, Wkh, Kh, M, KVc * HDc, Dc, KVc);
  gemm_wmma<1><<<dim3((KVc * HDc) / 128, M / 128), blk, 0, stream>>>(xh, Wvh, Vt, M, KVc * HDc, Dc, KVc);

  int nq = Bc * Hc  * Tc * (HDc / 2);
  int nk = Bc * KVc * Tc * (HDc / 2);
  rope_kernel<<<(nq + 255) / 256, 256, 0, stream>>>(Qh, nq);
  rope_kernel<<<(nk + 255) / 256, 256, 0, stream>>>(Kh, nk);

  attn_kernel<<<dim3(Tc / 64, Hc, Bc), dim3(128), 0, stream>>>(Qh, Kh, Vt, Oh);

  gemm_wmma<2><<<dim3(Dc / 128, M / 128), blk, 0, stream>>>(Oh, Woh, y, M, Dc, Hc * HDc, 0);
}
